// ASTGCN_block_32658931319128
// MI455X (gfx1250) — matrix operand order repaired and checked by reading, not compiled, not run on hardware
//
#include <hip/hip_runtime.h>

// ---- problem constants ----
#define NB   32
#define NN   900
#define FINC 16
#define TNUM 24
#define KCH  3
#define CCH  64     // C_CHEB
#define FT   64     // F_TIME
#define TB   2      // t-values per workgroup in kernel A (A-fragment reuse)

#define NPAD   912  // 57*16 padded n
#define MPAD   928  // 29*32 padded m (contraction)
#define NTILES 57
#define MSTEPS 29

// workspace layout (bytes, all 256-aligned)
#define CHEBT_OFF 0ull                    // 3*912*928*2      = 5,078,016
#define XH2_OFF   5078016ull              // 32*24*16*928*2   = 22,806,528
#define THP_OFF   27884544ull             // 64*64*2          = 8,192   (thetaT[c][kf])
#define WT_OFF    27892736ull             // 64*192*2         = 24,576  (WTt[c][K])
#define SH_OFF    27917312ull             // 32*900*24*64*2   = 88,473,600

typedef _Float16 v16h __attribute__((ext_vector_type(16)));
typedef _Float16 v8h  __attribute__((ext_vector_type(8)));
typedef _Float16 v2h  __attribute__((ext_vector_type(2)));
typedef float    v8f  __attribute__((ext_vector_type(8)));

__device__ __forceinline__ v8h ld8h(const _Float16* p) { return *(const v8h*)p; }
__device__ __forceinline__ v16h cat16(v8h a, v8h b) {
  return __builtin_shufflevector(a, b, 0,1,2,3,4,5,6,7,8,9,10,11,12,13,14,15);
}
__device__ __forceinline__ v16h ld16h(const _Float16* p) {          // 16 contiguous halfs
  return cat16(ld8h(p), ld8h(p + 8));
}
__device__ __forceinline__ v8f wmma_f16f32(v16h a, v16h b, v8f c) {
  // D(f32 16x16) = A(f16 16x32) x B(f16 32x16) + C
  return __builtin_amdgcn_wmma_f32_16x16x32_f16(false, a, false, b, (short)0, c, false, false);
}

// ---------------- prep: cheb (K,N,N) f32 -> chebT[k][n][m] f16, zero-padded ----------------
__global__ void prep_cheb(const float* __restrict__ cheb, _Float16* __restrict__ chebT) {
  int idx = blockIdx.x * 256 + threadIdx.x;
  int total = KCH * NPAD * MPAD;
  if (idx >= total) return;
  int k = idx / (NPAD * MPAD);
  int r = idx % (NPAD * MPAD);
  int n = r / MPAD;
  int m = r % MPAD;
  float v = 0.f;
  if (n < NN && m < NN) v = cheb[((size_t)k * NN + m) * NN + n];
  chebT[idx] = (_Float16)v;
}

// ---------------- prep: x (B,N,FIN,T) f32 -> xh2[b][t][f][m] f16 (m contiguous, padded) ----
__global__ void prep_x(const float* __restrict__ x, _Float16* __restrict__ xh2) {
  __shared__ float tile[32 * 384];          // 32 m-rows x (f*24+t)
  const int mblk = blockIdx.x;              // 0..28
  const int b    = blockIdx.y;              // 0..31
  const int tid  = threadIdx.x;             // 0..255
  for (int j = tid; j < 32 * 384; j += 256) {
    int ml = j / 384, ft = j % 384;
    int mg = mblk * 32 + ml;
    float v = 0.f;
    if (mg < NN) v = x[((size_t)b * NN + mg) * 384 + ft];   // (f,t) contiguous per (b,m)
    tile[j] = v;
  }
  __syncthreads();
  for (int j = tid; j < 32 * 384; j += 256) {
    int q = j / 32, ml = j % 32;            // q = t*16+f
    int t = q / 16, f = q % 16;
    xh2[(((size_t)b * TNUM + t) * FINC + f) * MPAD + mblk * 32 + ml] =
        (_Float16)tile[ml * 384 + f * TNUM + t];
  }
}

// -------- prep: thetaT[c][kf] (64x64) and WTt[c][K] (64x192) f16, K-contiguous per row --------
__global__ void prep_small(const float* __restrict__ theta, const float* __restrict__ time_w,
                           _Float16* __restrict__ thetaT, _Float16* __restrict__ WTt) {
  int idx = blockIdx.x * 256 + threadIdx.x;
  if (idx < 64 * 64) {
    int c = idx / 64, kf = idx % 64;        // kf = k*16+f, valid < 48
    float v = (kf < KCH * FINC) ? theta[(size_t)kf * CCH + c] : 0.f;
    thetaT[idx] = (_Float16)v;
  } else if (idx < 64 * 64 + 64 * 192) {
    int j = idx - 64 * 64;
    int c = j / 192, K = j % 192;           // K = dt*64 + cc
    int dt = K >> 6, cc = K & 63;
    WTt[j] = (_Float16)time_w[((size_t)c * CCH + cc) * 3 + dt];
  }
}

// ---------------- kernel A: cheb GEMM + theta GEMM + relu -> Sh[b][n][t][c] f16 ----------------
__global__ void gcn_kernel(const _Float16* __restrict__ chebT, const _Float16* __restrict__ xh2,
                           const _Float16* __restrict__ thetaT, _Float16* __restrict__ Sh) {
  const int nt = blockIdx.x;                // 0..56
  const int t0 = blockIdx.y * TB;           // 0,2,..,22
  const int b  = blockIdx.z;                // 0..31
  const int lane = threadIdx.x;             // 0..31 (one wave)
  const int lrow = lane & 15;
  const int sel  = lane >> 4;

  __shared__ float Rlds[16][48];
  __shared__ float Slds[16][64];

  // ---- stage 1: R_k[n,f] = sum_m chebT[k][n][m] * xh2[b][t][f][m], K-depth 928 ----
  const int nrow = nt * 16 + lrow;          // A row n
  const _Float16* arow[KCH];
#pragma unroll
  for (int k = 0; k < KCH; ++k) arow[k] = chebT + ((size_t)k * NPAD + nrow) * MPAD;
  const _Float16* xrow[TB];
#pragma unroll
  for (int tb = 0; tb < TB; ++tb)
    xrow[tb] = xh2 + (((size_t)b * TNUM + (t0 + tb)) * FINC + lrow) * MPAD;   // B col f = lrow

  v8f acc[KCH][TB] = {};

  // software pipeline: prefetch next m-step fragments before consuming current ones
  v16h bcur[TB], acur[KCH];
#pragma unroll
  for (int tb = 0; tb < TB; ++tb)
    bcur[tb] = ld16h(xrow[tb] + sel * 16);
#pragma unroll
  for (int k = 0; k < KCH; ++k)
    acur[k] = ld16h(arow[k] + sel * 16);

  for (int ms = 0; ms < MSTEPS - 1; ++ms) {
    const int mbn = (ms + 1) * 32;
    v16h bnxt[TB], anxt[KCH];
#pragma unroll
    for (int tb = 0; tb < TB; ++tb)
      bnxt[tb] = ld16h(xrow[tb] + mbn + sel * 16);
#pragma unroll
    for (int k = 0; k < KCH; ++k)
      anxt[k] = ld16h(arow[k] + mbn + sel * 16);
#pragma unroll
    for (int k = 0; k < KCH; ++k)
#pragma unroll
      for (int tb = 0; tb < TB; ++tb)
        acc[k][tb] = wmma_f16f32(acur[k], bcur[tb], acc[k][tb]);
#pragma unroll
    for (int tb = 0; tb < TB; ++tb) bcur[tb] = bnxt[tb];
#pragma unroll
    for (int k = 0; k < KCH; ++k) acur[k] = anxt[k];
  }
#pragma unroll
  for (int k = 0; k < KCH; ++k)
#pragma unroll
    for (int tb = 0; tb < TB; ++tb)
      acc[k][tb] = wmma_f16f32(acur[k], bcur[tb], acc[k][tb]);

  // ---- stage 2 (per t): out[n,c] = sum_{kf<48} R[n,kf] * thetaT[c][kf], K padded to 64 ----
  for (int tb = 0; tb < TB; ++tb) {
    __syncthreads();                        // protect Rlds/Slds reuse across tb
    // D layout: element (M = r + 8*sel, N = lrow)
#pragma unroll
    for (int k = 0; k < KCH; ++k)
#pragma unroll
      for (int r = 0; r < 8; ++r)
        Rlds[r + 8 * sel][k * 16 + lrow] = acc[k][tb][r];
    __syncthreads();

    v8f acc2[4] = {};
#pragma unroll
    for (int ks = 0; ks < 2; ++ks) {
      v16h a2;
#pragma unroll
      for (int i = 0; i < 16; ++i) {
        int K = ks * 32 + sel * 16 + i;
        a2[i] = (K < 48) ? (_Float16)Rlds[lrow][K] : (_Float16)0.f;
      }
#pragma unroll
      for (int ct = 0; ct < 4; ++ct) {
        v16h b2 = ld16h(thetaT + (ct * 16 + lrow) * 64 + ks * 32 + sel * 16);
        acc2[ct] = wmma_f16f32(a2, b2, acc2[ct]);
      }
    }
    __syncthreads();
#pragma unroll
    for (int ct = 0; ct < 4; ++ct)
#pragma unroll
      for (int r = 0; r < 8; ++r)
        Slds[r + 8 * sel][ct * 16 + lrow] = fmaxf(acc2[ct][r], 0.f);   // relu
    __syncthreads();

    // store Sh[b][n][t][c], lane packs 2 halfs (c = 2*lane, 2*lane+1)
    const int t = t0 + tb;
    for (int r0 = 0; r0 < 16; ++r0) {
      int n2 = nt * 16 + r0;
      if (n2 < NN) {
        _Float16* dst = Sh + (((size_t)b * NN + n2) * TNUM + t) * FT;
        v2h v;
        v[0] = (_Float16)Slds[r0][2 * lane];
        v[1] = (_Float16)Slds[r0][2 * lane + 1];
        *(v2h*)(dst + 2 * lane) = v;
      }
    }
  }
}

// ------- kernel B: time conv (WMMA) + residual 1x1 + bias + relu + LayerNorm + transpose -------
__global__ void tconv_ln_kernel(const _Float16* __restrict__ Sh, const _Float16* __restrict__ WTt,
                                const float* __restrict__ x, const float* __restrict__ res_w,
                                const float* __restrict__ res_b, const float* __restrict__ time_b,
                                const float* __restrict__ gamma, const float* __restrict__ beta,
                                float* __restrict__ out) {
  const int n = blockIdx.x;                 // 0..899
  const int b = blockIdx.y;                 // 0..31
  const int lane = threadIdx.x;
  const int lrow = lane & 15;
  const int sel  = lane >> 4;

  __shared__ float xs[FINC * TNUM];         // x[b][n][f][t]
  __shared__ float ybuf[TNUM][65];          // +1 pad to kill bank conflicts

  const float* xrow = x + ((size_t)b * NN + n) * (FINC * TNUM);
  for (int j = lane; j < FINC * TNUM; j += 32) xs[j] = xrow[j];
  __syncthreads();

  const _Float16* Sbase = Sh + ((size_t)b * NN + n) * TNUM * FT;   // [t][c]

  // tc[t,c] = sum_{K=dt*64+cc} S[t+dt-1][cc] * WTt[c][K], K-depth 192 = 6*32
  v8f acc[2][4] = {};
  for (int ks = 0; ks < 6; ++ks) {
    v16h a3[2];
#pragma unroll
    for (int tt = 0; tt < 2; ++tt) {
      const int trow = tt * 16 + lrow;
      v8h lo = {}, hi = {};
      {
        int K = ks * 32 + sel * 16;
        int dt = K >> 6, cc = K & 63;
        int tp = trow + dt - 1;
        if (tp >= 0 && tp < TNUM) lo = ld8h(Sbase + tp * FT + cc);
      }
      {
        int K = ks * 32 + sel * 16 + 8;
        int dt = K >> 6, cc = K & 63;
        int tp = trow + dt - 1;
        if (tp >= 0 && tp < TNUM) hi = ld8h(Sbase + tp * FT + cc);
      }
      a3[tt] = cat16(lo, hi);
    }
#pragma unroll
    for (int ct = 0; ct < 4; ++ct) {
      v16h b3 = ld16h(WTt + (ct * 16 + lrow) * 192 + ks * 32 + sel * 16);
      acc[0][ct] = wmma_f16f32(a3[0], b3, acc[0][ct]);
      acc[1][ct] = wmma_f16f32(a3[1], b3, acc[1][ct]);
    }
  }

  // epilogue: bias + residual 1x1 conv + relu into ybuf[t][c]
#pragma unroll
  for (int ct = 0; ct < 4; ++ct) {
    const int c = ct * 16 + lrow;
    float rw[FINC];
#pragma unroll
    for (int f = 0; f < FINC; ++f) rw[f] = res_w[c * FINC + f];
    const float tb = time_b[c];
    const float rb = res_b[c];
#pragma unroll
    for (int tt = 0; tt < 2; ++tt)
#pragma unroll
      for (int r = 0; r < 8; ++r) {
        int t = tt * 16 + r + 8 * sel;
        if (t < TNUM) {
          float rc = rb;
#pragma unroll
          for (int f = 0; f < FINC; ++f) rc += rw[f] * xs[f * TNUM + t];
          ybuf[t][c] = fmaxf(acc[tt][ct][r] + tb + rc, 0.f);
        }
      }
  }
  __syncthreads();

  // LayerNorm over 64 channels per t (lane = t), then store out[b][n][c][t] (t coalesced)
  float mu = 0.f, rs = 0.f;
  if (lane < TNUM) {
    float s = 0.f;
    for (int c = 0; c < FT; ++c) s += ybuf[lane][c];
    mu = s * (1.0f / 64.0f);
    float s2 = 0.f;
    for (int c = 0; c < FT; ++c) { float d = ybuf[lane][c] - mu; s2 += d * d; }
    rs = rsqrtf(s2 * (1.0f / 64.0f) + 1e-5f);
  }
  float* obase = out + ((size_t)b * NN + n) * FT * TNUM;
  for (int c = 0; c < FT; ++c) {
    if (lane < TNUM) {
      obase[c * TNUM + lane] = (ybuf[lane][c] - mu) * rs * gamma[c] + beta[c];
    }
  }
}

extern "C" void kernel_launch(void* const* d_in, const int* in_sizes, int n_in,
                              void* d_out, int out_size, void* d_ws, size_t ws_size,
                              hipStream_t stream) {
  const float* x      = (const float*)d_in[0];
  const float* cheb   = (const float*)d_in[1];
  const float* theta  = (const float*)d_in[2];
  const float* time_w = (const float*)d_in[3];
  const float* time_b = (const float*)d_in[4];
  const float* res_w  = (const float*)d_in[5];
  const float* res_b  = (const float*)d_in[6];
  const float* gamma  = (const float*)d_in[7];
  const float* beta   = (const float*)d_in[8];
  float* out = (float*)d_out;

  char* ws = (char*)d_ws;
  _Float16* chebT  = (_Float16*)(ws + CHEBT_OFF);
  _Float16* xh2    = (_Float16*)(ws + XH2_OFF);
  _Float16* thetaT = (_Float16*)(ws + THP_OFF);
  _Float16* WTt    = (_Float16*)(ws + WT_OFF);
  _Float16* Sh     = (_Float16*)(ws + SH_OFF);

  // prep passes (f32 -> f16, transposed/padded operand layouts)
  {
    int total = KCH * NPAD * MPAD;
    prep_cheb<<<(total + 255) / 256, 256, 0, stream>>>(cheb, chebT);
  }
  prep_x<<<dim3(MSTEPS, NB), 256, 0, stream>>>(x, xh2);
  prep_small<<<64, 256, 0, stream>>>(theta, time_w, thetaT, WTt);

  // fused cheb-GEMM (K=928, double-buffered, TB=2 t-reuse) + theta-GEMM (K=64) + relu
  gcn_kernel<<<dim3(NTILES, TNUM / TB, NB), 32, 0, stream>>>(chebT, xh2, thetaT, Sh);

  // time conv GEMM (K=192) + residual + relu + LayerNorm + output transpose
  tconv_ln_kernel<<<dim3(NN, NB), 32, 0, stream>>>(Sh, WTt, x, res_w, res_b,
                                                   time_b, gamma, beta, out);
}
